// SlidingWindowMultiHeadAttention_18339510354077
// MI455X (gfx1250) — matrix operand order repaired and checked by reading, not compiled, not run on hardware
//
#include <hip/hip_runtime.h>
#include <hip/hip_bf16.h>
#include <math.h>

typedef __bf16 bf16;
typedef __attribute__((ext_vector_type(16))) __bf16 v16bf;
typedef __attribute__((ext_vector_type(8)))  __bf16 v8bf;
typedef __attribute__((ext_vector_type(8)))  float  v8f;

#define WMMA_BF16(a, b, c) \
  __builtin_amdgcn_wmma_f32_16x16x32_bf16(false, (a), false, (b), (short)0, (c), false, false)

static constexpr int   BB   = 2;
static constexpr int   S    = 2048;
static constexpr int   DIN  = 1024;
static constexpr int   DOUT = 1024;
static constexpr int   H    = 16;
static constexpr int   HD   = 64;      // head dim
static constexpr int   WIN  = 256;     // band half-width
static constexpr float SCALE = 0.125f; // 1/sqrt(64)

// ---------------------------------------------------------------------------
// WMMA operand loaders following CDNA5 ISA 7.12.2 VGPR layouts (wave32).
// A (16x32 bf16, M x K): lane l(0..15)+half*16 holds row M=l;
//   elems 0..7  = K[half*8    .. half*8+7]    (contiguous, 16B)
//   elems 8..15 = K[16+half*8 .. 16+half*8+7] (contiguous, 16B)
// B (32x16 bf16, K x N): lane holds col N=l; elems 0..7 = K[half*8 .. half*8+7],
//   elems 8..15 = K[16+half*8 .. 16+half*8+7] (same K order as A).
// C/D (16x16 f32): element r <-> (row = r + half*8, col = l).
// ---------------------------------------------------------------------------
__device__ __forceinline__ v16bf load_A_tile(const bf16* __restrict__ base,
                                             int lda, int row0, int k0) {
  const int lane = threadIdx.x & 31;
  const int l = lane & 15, half = lane >> 4;
  const bf16* p = base + (size_t)(row0 + l) * lda + k0 + half * 8;
  union { v16bf v; v8bf h[2]; } u;
  u.h[0] = *(const v8bf*)(p);
  u.h[1] = *(const v8bf*)(p + 16);
  return u.v;
}

// src is N-major with leading dim ldn; B[k][n] = src[(col0+n)*ldn + k0 + k]
__device__ __forceinline__ v16bf load_B_tile(const bf16* __restrict__ base,
                                             int ldn, int col0, int k0) {
  const int lane = threadIdx.x & 31;
  const int l = lane & 15, half = lane >> 4;
  const bf16* p = base + (size_t)(col0 + l) * ldn + k0 + half * 8;
  union { v16bf v; v8bf h[2]; } u;
  u.h[0] = *(const v8bf*)(p);
  u.h[1] = *(const v8bf*)(p + 16);
  return u.v;
}

// ---------------------------------------------------------------------------
// Stage 0: fp32 -> bf16 conversion
// ---------------------------------------------------------------------------
__global__ void cvt_f32_bf16_kernel(const float* __restrict__ src,
                                    bf16* __restrict__ dst, int n) {
  int i = blockIdx.x * blockDim.x + threadIdx.x;
  int stride = gridDim.x * blockDim.x;
  for (; i < n; i += stride) dst[i] = (bf16)src[i];
}

// ---------------------------------------------------------------------------
// Stage 1: QKV projection.  C[m,n] = sum_k X[m,k] * Wqkv[n,k]
// Register-blocked 16x64 per wave (A reused across 4 B tiles).
// Ping-pong double buffering (k-loop unrolled by 2, buffers refill in
// place): no cross-buffer register copies, loads stay ~10 deep in flight
// against the other buffer's WMMAs.
// Results scattered into head-split layouts:
//   Q,K : [b,h,s,64] bf16;  Vt : [b,h,64,s] bf16 (pre-transposed).
// ---------------------------------------------------------------------------
__global__ __launch_bounds__(256) void qkv_gemm_kernel(
    const bf16* __restrict__ X, const bf16* __restrict__ Wqkv,
    bf16* __restrict__ Q, bf16* __restrict__ Kk, bf16* __restrict__ Vt) {
  const int wave = threadIdx.x >> 5;
  const int lane = threadIdx.x & 31;
  const int l = lane & 15, half = lane >> 4;
  const int NB = (3 * DOUT) / 64;  // 48 column blocks
  const int blk = blockIdx.x * 8 + wave;
  const int row0 = (blk / NB) * 16;
  const int col0 = (blk % NB) * 64;

  v8f acc[4] = {};
  v16bf a0 = load_A_tile(X, DIN, row0, 0);
  v16bf b0[4];
#pragma unroll
  for (int t = 0; t < 4; ++t) b0[t] = load_B_tile(Wqkv, DIN, col0 + t * 16, 0);
  v16bf a1 = load_A_tile(X, DIN, row0, 32);
  v16bf b1[4];
#pragma unroll
  for (int t = 0; t < 4; ++t) b1[t] = load_B_tile(Wqkv, DIN, col0 + t * 16, 32);

  for (int k0 = 64; k0 < DIN; k0 += 64) {
#pragma unroll
    for (int t = 0; t < 4; ++t) acc[t] = WMMA_BF16(a0, b0[t], acc[t]);
    a0 = load_A_tile(X, DIN, row0, k0);
#pragma unroll
    for (int t = 0; t < 4; ++t)
      b0[t] = load_B_tile(Wqkv, DIN, col0 + t * 16, k0);
#pragma unroll
    for (int t = 0; t < 4; ++t) acc[t] = WMMA_BF16(a1, b1[t], acc[t]);
    a1 = load_A_tile(X, DIN, row0, k0 + 32);
#pragma unroll
    for (int t = 0; t < 4; ++t)
      b1[t] = load_B_tile(Wqkv, DIN, col0 + t * 16, k0 + 32);
  }
#pragma unroll
  for (int t = 0; t < 4; ++t) acc[t] = WMMA_BF16(a0, b0[t], acc[t]);
#pragma unroll
  for (int t = 0; t < 4; ++t) acc[t] = WMMA_BF16(a1, b1[t], acc[t]);

  const int which = col0 / DOUT;        // 0=q 1=k 2=v (uniform per block)
  const int h = (col0 % DOUT) / HD;     // uniform per block
#pragma unroll
  for (int t = 0; t < 4; ++t) {
    const int hd = t * 16 + l;          // col0 is 64-aligned
#pragma unroll
    for (int r = 0; r < 8; ++r) {
      const int m = row0 + r + half * 8;
      const int bb = m / S, s = m % S;
      const bf16 val = (bf16)acc[t][r];
      if (which == 0)
        Q[((size_t)(bb * H + h) * S + s) * HD + hd] = val;
      else if (which == 1)
        Kk[((size_t)(bb * H + h) * S + s) * HD + hd] = val;
      else
        Vt[((size_t)(bb * H + h) * HD + hd) * S + s] = val;
    }
  }
}

// ---------------------------------------------------------------------------
// Stage 2: banded flash attention.  One wave per (b,h, 16-query tile).
// Scores: 4 WMMA / 32-key block; P restaged via per-wave LDS slab (C->A
// relayout); P@V: 4 WMMA / block into 16x64 f32 accumulator; online softmax.
// ---------------------------------------------------------------------------
__global__ __launch_bounds__(256) void attn_kernel(
    const bf16* __restrict__ Q, const bf16* __restrict__ K,
    const bf16* __restrict__ Vt, bf16* __restrict__ Ctx) {
  __shared__ __align__(16) bf16 lds[8][16][32];  // 1KB P-staging slab / wave

  const int wave = threadIdx.x >> 5;
  const int lane = threadIdx.x & 31;
  const int l = lane & 15, half = lane >> 4;

  const int task = blockIdx.x * 8 + wave;   // 4096 tasks
  const int qt = task & 127;                // 128 query tiles per (b,h)
  const int bh = task >> 7;                 // 0..31
  const int i0 = qt * 16;

  const bf16* Qp = Q  + (size_t)bh * S * HD;
  const bf16* Kp = K  + (size_t)bh * S * HD;
  const bf16* Vp = Vt + (size_t)bh * HD * S;

  // Q tile (16x64) kept in registers as two A operands (K=0..31, 32..63)
  const v16bf aq0 = load_A_tile(Qp, HD, i0, 0);
  const v16bf aq1 = load_A_tile(Qp, HD, i0, 32);

  float mrow[8], lrow[8];
  v8f o[4] = {};
#pragma unroll
  for (int r = 0; r < 8; ++r) { mrow[r] = -INFINITY; lrow[r] = 0.0f; }

  int jlo = i0 - WIN;      if (jlo < 0) jlo = 0;
  int jhi = i0 + 15 + WIN; if (jhi > S - 1) jhi = S - 1;
  const int js = jlo & ~31;

  for (int j0 = js; j0 <= jhi; j0 += 32) {
    // ---- scores: two 16x16 C tiles covering keys j0..j0+31 ----
    v8f c0 = {}, c1 = {};
    c0 = WMMA_BF16(aq0, load_B_tile(Kp, HD, j0, 0), c0);
    c0 = WMMA_BF16(aq1, load_B_tile(Kp, HD, j0, 32), c0);
    c1 = WMMA_BF16(aq0, load_B_tile(Kp, HD, j0 + 16, 0), c1);
    c1 = WMMA_BF16(aq1, load_B_tile(Kp, HD, j0 + 16, 32), c1);

    // ---- mask + online softmax (row M lives in one VGPR across 16 lanes) ----
#pragma unroll
    for (int r = 0; r < 8; ++r) {
      const int i = i0 + r + half * 8;
      int da = i - (j0 + l);        if (da < 0) da = -da;
      int db = i - (j0 + 16 + l);   if (db < 0) db = -db;
      float s0 = (da <= WIN) ? c0[r] * SCALE : -INFINITY;
      float s1 = (db <= WIN) ? c1[r] * SCALE : -INFINITY;

      float mx = fmaxf(s0, s1);
#pragma unroll
      for (int off = 1; off < 16; off <<= 1) mx = fmaxf(mx, __shfl_xor(mx, off));
      const float mn = fmaxf(mrow[r], mx);
      const float corr = __expf(mrow[r] - mn);
      const float e0 = __expf(s0 - mn);
      const float e1 = __expf(s1 - mn);
      float rs = e0 + e1;
#pragma unroll
      for (int off = 1; off < 16; off <<= 1) rs += __shfl_xor(rs, off);
      lrow[r] = lrow[r] * corr + rs;
      mrow[r] = mn;
#pragma unroll
      for (int t = 0; t < 4; ++t) o[t][r] *= corr;

      // C-layout -> LDS (row-major 16x32 P tile)
      lds[wave][r + half * 8][l]      = (bf16)e0;
      lds[wave][r + half * 8][16 + l] = (bf16)e1;
    }

    // ---- P (A-layout from LDS) @ V (B from pre-transposed Vt) ----
    const v16bf pa = load_A_tile(&lds[wave][0][0], 32, 0, 0);
#pragma unroll
    for (int t = 0; t < 4; ++t)
      o[t] = WMMA_BF16(pa, load_B_tile(Vp, S, t * 16, j0), o[t]);
  }

  // ---- normalize and store context, layout [b*s, 1024] bf16 ----
  const int bb = bh >> 4, hh = bh & 15;
#pragma unroll
  for (int t = 0; t < 4; ++t)
#pragma unroll
    for (int r = 0; r < 8; ++r) {
      const int row = bb * S + i0 + r + half * 8;
      const int col = hh * HD + t * 16 + l;
      Ctx[(size_t)row * DOUT + col] = (bf16)(o[t][r] / lrow[r]);
    }
}

// ---------------------------------------------------------------------------
// Stage 3: output projection + bias, register-blocked 16x64 per wave with
// the same ping-pong double buffering as the QKV GEMM.
// out[m,n] = sum_k ctx[m,k]*Wout[n,k] + b[n]
// ---------------------------------------------------------------------------
__global__ __launch_bounds__(256) void out_gemm_kernel(
    const bf16* __restrict__ Ctx, const bf16* __restrict__ Wout,
    const float* __restrict__ bias, float* __restrict__ Out) {
  const int wave = threadIdx.x >> 5;
  const int lane = threadIdx.x & 31;
  const int l = lane & 15, half = lane >> 4;
  const int NB = DOUT / 64;  // 16 column blocks
  const int blk = blockIdx.x * 8 + wave;
  const int row0 = (blk / NB) * 16;
  const int col0 = (blk % NB) * 64;

  v8f acc[4] = {};
  v16bf a0 = load_A_tile(Ctx, DOUT, row0, 0);
  v16bf b0[4];
#pragma unroll
  for (int t = 0; t < 4; ++t) b0[t] = load_B_tile(Wout, DOUT, col0 + t * 16, 0);
  v16bf a1 = load_A_tile(Ctx, DOUT, row0, 32);
  v16bf b1[4];
#pragma unroll
  for (int t = 0; t < 4; ++t) b1[t] = load_B_tile(Wout, DOUT, col0 + t * 16, 32);

  for (int k0 = 64; k0 < DOUT; k0 += 64) {
#pragma unroll
    for (int t = 0; t < 4; ++t) acc[t] = WMMA_BF16(a0, b0[t], acc[t]);
    a0 = load_A_tile(Ctx, DOUT, row0, k0);
#pragma unroll
    for (int t = 0; t < 4; ++t)
      b0[t] = load_B_tile(Wout, DOUT, col0 + t * 16, k0);
#pragma unroll
    for (int t = 0; t < 4; ++t) acc[t] = WMMA_BF16(a1, b1[t], acc[t]);
    a1 = load_A_tile(Ctx, DOUT, row0, k0 + 32);
#pragma unroll
    for (int t = 0; t < 4; ++t)
      b1[t] = load_B_tile(Wout, DOUT, col0 + t * 16, k0 + 32);
  }
#pragma unroll
  for (int t = 0; t < 4; ++t) acc[t] = WMMA_BF16(a0, b0[t], acc[t]);
#pragma unroll
  for (int t = 0; t < 4; ++t) acc[t] = WMMA_BF16(a1, b1[t], acc[t]);

#pragma unroll
  for (int t = 0; t < 4; ++t) {
    const int n = col0 + t * 16 + l;
    const float bv = bias[n];
#pragma unroll
    for (int r = 0; r < 8; ++r)
      Out[(size_t)(row0 + r + half * 8) * DOUT + n] = acc[t][r] + bv;
  }
}

// ---------------------------------------------------------------------------
extern "C" void kernel_launch(void* const* d_in, const int* in_sizes, int n_in,
                              void* d_out, int out_size, void* d_ws, size_t ws_size,
                              hipStream_t stream) {
  const float* x    = (const float*)d_in[0];  // [2, 2048, 1024]
  const float* Wqkv = (const float*)d_in[1];  // [3072, 1024]
  const float* Wout = (const float*)d_in[2];  // [1024, 1024]
  const float* bout = (const float*)d_in[3];  // [1024]
  float* out = (float*)d_out;                 // [2, 2048, 1024] f32

  char* ws = (char*)d_ws;
  size_t off = 0;
  bf16* xbf    = (bf16*)(ws + off); off += (size_t)BB * S * DIN * 2;       // 8 MB
  bf16* wqkvbf = (bf16*)(ws + off); off += (size_t)3 * DOUT * DIN * 2;     // 6 MB
  bf16* woutbf = (bf16*)(ws + off); off += (size_t)DOUT * DOUT * 2;        // 2 MB
  bf16* Qh     = (bf16*)(ws + off); off += (size_t)BB * H * S * HD * 2;    // 8 MB
  bf16* Kh     = (bf16*)(ws + off); off += (size_t)BB * H * S * HD * 2;    // 8 MB
  bf16* Vth    = (bf16*)(ws + off); off += (size_t)BB * H * HD * S * 2;    // 8 MB
  bf16* Ctx    = (bf16*)(ws + off); off += (size_t)BB * S * DOUT * 2;      // 8 MB

  cvt_f32_bf16_kernel<<<2048, 256, 0, stream>>>(x, xbf, BB * S * DIN);
  cvt_f32_bf16_kernel<<<2048, 256, 0, stream>>>(Wqkv, wqkvbf, 3 * DOUT * DIN);
  cvt_f32_bf16_kernel<<<1024, 256, 0, stream>>>(Wout, woutbf, DOUT * DOUT);

  // 256 row tiles x 48 column blocks = 12288 waves / 8 per block
  qkv_gemm_kernel<<<1536, 256, 0, stream>>>(xbf, wqkvbf, Qh, Kh, Vth);

  // 2*16*128 = 4096 wave tasks / 8 per block
  attn_kernel<<<512, 256, 0, stream>>>(Qh, Kh, Vth, Ctx);

  // 256 row tiles x 16 column blocks = 4096 waves / 8 per block
  out_gemm_kernel<<<512, 256, 0, stream>>>(Ctx, woutbf, bout, out);
}
